// MultiHeadSpikingAttention_64098091926007
// MI455X (gfx1250) — 16-bit and 8-bit matrix operand order repaired and checked by reading, compiled, not run on hardware
//
#include <hip/hip_runtime.h>
#include <hip/hip_bf16.h>


typedef __attribute__((ext_vector_type(16))) _Float16 v16h;
typedef __attribute__((ext_vector_type(8)))  float    v8f;
typedef __attribute__((ext_vector_type(8)))  int      v8i;
typedef __attribute__((ext_vector_type(4)))  unsigned uv4;
typedef __attribute__((ext_vector_type(2)))  unsigned uv2;

union F16Frag { v16h v; uv4 q[2]; unsigned u[8]; _Float16 h[16]; };
union I8Frag  { v8i  v; uv4 q[2]; uv2 d[4]; };
union AccI    { v8i  v; int   i[8]; };
union AccF    { v8f  v; float f[8]; };

#define S_LEN  2048
#define D_LEN  1024
#define M_LEN  4096
#define HD     64
#define K_STEP 32
#define NITER  (D_LEN / K_STEP)

#if __has_builtin(__builtin_amdgcn_global_load_async_to_lds_b128)
#define HAVE_ASYNC 1
#else
#define HAVE_ASYNC 0
#endif

template <int N>
__device__ __forceinline__ void wait_async() {
#if __has_builtin(__builtin_amdgcn_s_wait_asynccnt)
  __builtin_amdgcn_s_wait_asynccnt(N);
#else
  asm volatile("s_wait_asynccnt %0" ::"i"(N) : "memory");
#endif
}

#if HAVE_ASYNC
typedef __attribute__((vector_size(16))) int i32x4;
typedef __attribute__((address_space(1))) i32x4 gi32x4;
typedef __attribute__((address_space(3))) i32x4 li32x4;

__device__ __forceinline__ void async_copy_b128(const _Float16* g, _Float16* l) {
  __builtin_amdgcn_global_load_async_to_lds_b128(
      (gi32x4*)(i32x4*)(void*)g,
      (li32x4*)(i32x4*)(void*)l, 0, 0);
}
#endif

__global__ void cvt_f32_to_f16(const float* __restrict__ src,
                               _Float16* __restrict__ dst, int n) {
  int i = blockIdx.x * blockDim.x + threadIdx.x;
  if (i < n) dst[i] = (_Float16)src[i];
}

__global__ void cvt_w_interleave(const float* __restrict__ src,
                                 _Float16* __restrict__ dst, int n) {
  int i = blockIdx.x * blockDim.x + threadIdx.x;
  if (i < n) {
    int k = i >> 10, nn = i & 1023;
    dst[(size_t)((k >> 1) * 1024 + nn) * 2 + (k & 1)] = (_Float16)src[i];
  }
}

__device__ __forceinline__ void stage_A(const _Float16* __restrict__ Ag,
                                        int m0, int k0,
                                        _Float16* __restrict__ dst) {
  const int t = threadIdx.x;
  const int row = t >> 1, part = t & 1;
  const _Float16* g = Ag + (size_t)(m0 + row) * D_LEN + k0 + part * 16;
  _Float16* l = dst + row * 32 + part * 16;
#if HAVE_ASYNC
  async_copy_b128(g, l);
  async_copy_b128(g + 8, l + 8);
#else
  const uv4* s = (const uv4*)g;
  uv4 a0 = s[0], a1 = s[1];
  uv4* d = (uv4*)l;
  d[0] = a0; d[1] = a1;
#endif
}

__device__ __forceinline__ void gemm_core(const _Float16* __restrict__ Ag,
                                          const unsigned* __restrict__ Bpi,
                                          int m0, int n0, v8f acc[4]) {
  __shared__ _Float16 As[2][128 * 32];

  const int tid  = threadIdx.x;
  const int lane = tid & 31, wave = tid >> 5;
  const int wm = wave >> 1, wn = wave & 1;
  const int r = lane & 15, hi = lane >> 4;

  acc[0] = (v8f){}; acc[1] = (v8f){}; acc[2] = (v8f){}; acc[3] = (v8f){};

  stage_A(Ag, m0, 0, As[0]);

  for (int i = 0; i < NITER; ++i) {
    const int cur = i & 1;
    if (i + 1 < NITER) {
      stage_A(Ag, m0, (i + 1) * K_STEP, As[cur ^ 1]);
#if HAVE_ASYNC
      wait_async<2>();
#endif
    } else {
#if HAVE_ASYNC
      wait_async<0>();
#endif
    }
    __syncthreads();

    F16Frag af[2];
#pragma unroll
    for (int ii = 0; ii < 2; ++ii) {
      const uv4* ap = (const uv4*)(As[cur] + (wm * 32 + ii * 16 + r) * 32 + hi * 16);
      af[ii].q[0] = ap[0];
      af[ii].q[1] = ap[1];
    }
    F16Frag bf[2];
    const unsigned* Bp = Bpi + (size_t)(i * 16 + hi * 8) * D_LEN + n0 + wn * 32 + r;
#pragma unroll
    for (int j = 0; j < 2; ++j)
#pragma unroll
      for (int v = 0; v < 8; ++v)
        bf[j].u[v] = Bp[(size_t)v * D_LEN + j * 16];

#pragma unroll
    for (int ii = 0; ii < 2; ++ii)
#pragma unroll
      for (int j = 0; j < 2; ++j)
        acc[ii * 2 + j] = __builtin_amdgcn_wmma_f32_16x16x32_f16(
            false, af[ii].v, false, bf[j].v, (short)0, acc[ii * 2 + j], false, false);

    __syncthreads();
  }
}

__global__ __launch_bounds__(256)
void proj_lif_kernel(const _Float16* __restrict__ x16,
                     const unsigned* __restrict__ w16pi,
                     const float* __restrict__ qb, const float* __restrict__ kb,
                     const float* __restrict__ vb,
                     unsigned char* __restrict__ qk4base,
                     _Float16* __restrict__ v16sw) {
  const int z = blockIdx.z;
  const unsigned* W = w16pi + (size_t)z * (D_LEN * D_LEN / 2);
  const float* bias = (z == 0) ? qb : ((z == 1) ? kb : vb);
  const int m0 = blockIdx.y * 128, n0 = blockIdx.x * 64;

  v8f acc[4];
  gemm_core(x16, W, m0, n0, acc);

  const int lane = threadIdx.x & 31, wave = threadIdx.x >> 5;
  const int wm = wave >> 1, wn = wave & 1;
  const int r = lane & 15, hi = lane >> 4;

#pragma unroll
  for (int i = 0; i < 2; ++i)
#pragma unroll
    for (int j = 0; j < 2; ++j) {
      AccF u; u.v = acc[i * 2 + j];
#pragma unroll
      for (int e = 0; e < 8; ++e) {
        int gm = m0 + wm * 32 + i * 16 + 8 * hi + e;
        int gn = n0 + wn * 32 + j * 16 + r;
        float cur = u.f[e] + bias[gn];
        float mem = 0.f, cnt = 0.f;
#pragma unroll
        for (int t = 0; t < 4; ++t) {
          float reset = (mem > 1.0f) ? 1.0f : 0.0f;
          mem = 0.95f * mem + cur - reset;
          cnt += (mem > 1.0f) ? 1.0f : 0.0f;
        }
        int b = gm >> 11, s = gm & (S_LEN - 1);
        int h = gn >> 6,  hd = gn & (HD - 1);
        int bh = b * 16 + h;
        if (z < 2) {
          qk4base[(size_t)z * 4194304 + ((size_t)bh * S_LEN + s) * HD + hd] =
              (unsigned char)(int)cnt;
        } else {
          size_t idx = (((size_t)bh * (S_LEN / 2) + (s >> 1)) * HD + hd) * 2 + (s & 1);
          v16sw[idx] = (_Float16)(cnt * 0.25f);
        }
      }
    }
}

__global__ __launch_bounds__(256)
void attn_kernel(const unsigned char* __restrict__ q4,
                 const unsigned char* __restrict__ k4,
                 const unsigned* __restrict__ v16sw,
                 _Float16* __restrict__ attn16) {
  __shared__ _Float16 Plds[8 * 512];

  const int tid = threadIdx.x;
  const int lane = tid & 31, wave = tid >> 5;
  _Float16* Pw = Plds + wave * 512;

  const int bh = blockIdx.y;
  const int b = bh >> 4, h = bh & 15;
  const int q0 = (blockIdx.x * 8 + wave) * 16;
  const int r = lane & 15, hi = lane >> 4;

  const unsigned char* qbh = q4 + (size_t)bh * (S_LEN * HD);
  const unsigned char* kbh = k4 + (size_t)bh * (S_LEN * HD);
  const unsigned*      vbh = v16sw + (size_t)bh * (S_LEN / 2) * HD;

  I8Frag aq;
#pragma unroll
  for (int g = 0; g < 4; ++g)
    aq.d[g] = *(const uv2*)(qbh + (q0 + r) * HD + (g >> 1) * 32 + (g & 1) * 8 + hi * 16);

  v8f oacc[4]; oacc[0] = (v8f){}; oacc[1] = (v8f){}; oacc[2] = (v8f){}; oacc[3] = (v8f){};
  float den[8];
#pragma unroll
  for (int j = 0; j < 8; ++j) den[j] = 0.f;

  for (int it = 0; it < S_LEN / 32; ++it) {
    const int kk = it * 32;
    if (it + 1 < S_LEN / 32) {
      __builtin_prefetch(kbh + (kk + 32 + lane) * HD, 0, 0);
      __builtin_prefetch(vbh + (size_t)((it + 1) * 16 + r) * HD, 0, 0);
    }
    I8Frag bk[2];
#pragma unroll
    for (int f = 0; f < 2; ++f) {
      const uv4* kp = (const uv4*)(kbh + (kk + f * 16 + r) * HD + hi * 16);
      bk[f].q[0] = kp[0];
      bk[f].q[1] = kp[2];
    }
    v8i zi = {};
    AccI s0, s1;
    s0.v = __builtin_amdgcn_wmma_i32_16x16x64_iu8(false, aq.v, false, bk[0].v, zi, false, false);
    s1.v = __builtin_amdgcn_wmma_i32_16x16x64_iu8(false, aq.v, false, bk[1].v, zi, false, false);

    float p0[8], p1[8];
#pragma unroll
    for (int j = 0; j < 8; ++j) {
      p0[j] = __expf((float)s0.i[j] * 0.0078125f);
      p1[j] = __expf((float)s1.i[j] * 0.0078125f);
    }
#pragma unroll
    for (int j = 0; j < 8; ++j) {
      float t = p0[j] + p1[j];
      t += __shfl_xor(t, 1, 32);
      t += __shfl_xor(t, 2, 32);
      t += __shfl_xor(t, 4, 32);
      t += __shfl_xor(t, 8, 32);
      den[j] += t;
    }
#pragma unroll
    for (int j = 0; j < 8; ++j) {
      Pw[(j + 8 * hi) * 32 + r]      = (_Float16)p0[j];
      Pw[(j + 8 * hi) * 32 + 16 + r] = (_Float16)p1[j];
    }
    F16Frag pf;
    {
      const uv4* pp = (const uv4*)(Pw + r * 32 + hi * 16);
      pf.q[0] = pp[0];
      pf.q[1] = pp[1];
    }
#pragma unroll
    for (int dc = 0; dc < 4; ++dc) {
      F16Frag vf;
#pragma unroll
      for (int v = 0; v < 8; ++v)
        vf.u[v] = vbh[(size_t)(it * 16 + hi * 8 + v) * HD + dc * 16 + r];
      oacc[dc] = __builtin_amdgcn_wmma_f32_16x16x32_f16(
          false, pf.v, false, vf.v, (short)0, oacc[dc], false, false);
    }
  }

  float inv[8];
#pragma unroll
  for (int j = 0; j < 8; ++j) inv[j] = 1.0f / den[j];
#pragma unroll
  for (int dc = 0; dc < 4; ++dc) {
    AccF o; o.v = oacc[dc];
#pragma unroll
    for (int j = 0; j < 8; ++j) {
      int row = q0 + j + 8 * hi;
      int col = h * HD + dc * 16 + r;
      attn16[((size_t)b * S_LEN + row) * D_LEN + col] = (_Float16)(o.f[j] * inv[j]);
    }
  }
}

__global__ __launch_bounds__(256)
void out_gemm_kernel(const _Float16* __restrict__ a16,
                     const unsigned* __restrict__ ow16pi,
                     const float* __restrict__ ob,
                     float* __restrict__ dout) {
  const int m0 = blockIdx.y * 128, n0 = blockIdx.x * 64;
  v8f acc[4];
  gemm_core(a16, ow16pi, m0, n0, acc);

  const int lane = threadIdx.x & 31, wave = threadIdx.x >> 5;
  const int wm = wave >> 1, wn = wave & 1;
  const int r = lane & 15, hi = lane >> 4;
#pragma unroll
  for (int i = 0; i < 2; ++i)
#pragma unroll
    for (int j = 0; j < 2; ++j) {
      AccF u; u.v = acc[i * 2 + j];
#pragma unroll
      for (int e = 0; e < 8; ++e) {
        int gm = m0 + wm * 32 + i * 16 + 8 * hi + e;
        int gn = n0 + wn * 32 + j * 16 + r;
        dout[(size_t)gm * D_LEN + gn] = u.f[e] + ob[gn];
      }
    }
}

#define OFF_X16   0u
#define OFF_W16   8388608u
#define OFF_OW16  14680064u
#define OFF_Q4    16777216u
#define OFF_K4    20971520u
#define OFF_V16   25165824u
#define OFF_AT16  33554432u

extern "C" void kernel_launch(void* const* d_in, const int* in_sizes, int n_in,
                              void* d_out, int out_size, void* d_ws, size_t ws_size,
                              hipStream_t stream) {
  const float* x  = (const float*)d_in[0];
  const float* qw = (const float*)d_in[1];
  const float* qb = (const float*)d_in[2];
  const float* kw = (const float*)d_in[3];
  const float* kb = (const float*)d_in[4];
  const float* vw = (const float*)d_in[5];
  const float* vb = (const float*)d_in[6];
  const float* ow = (const float*)d_in[7];
  const float* ob = (const float*)d_in[8];

  char* ws = (char*)d_ws;
  _Float16* x16  = (_Float16*)(ws + OFF_X16);
  _Float16* w16  = (_Float16*)(ws + OFF_W16);
  _Float16* ow16 = (_Float16*)(ws + OFF_OW16);

  cvt_f32_to_f16 <<<M_LEN * D_LEN / 256, 256, 0, stream>>>(x, x16, M_LEN * D_LEN);
  cvt_w_interleave<<<D_LEN * D_LEN / 256, 256, 0, stream>>>(qw, w16,               D_LEN * D_LEN);
  cvt_w_interleave<<<D_LEN * D_LEN / 256, 256, 0, stream>>>(kw, w16 + 1048576,     D_LEN * D_LEN);
  cvt_w_interleave<<<D_LEN * D_LEN / 256, 256, 0, stream>>>(vw, w16 + 2 * 1048576, D_LEN * D_LEN);
  cvt_w_interleave<<<D_LEN * D_LEN / 256, 256, 0, stream>>>(ow, ow16,              D_LEN * D_LEN);

  proj_lif_kernel<<<dim3(D_LEN / 64, M_LEN / 128, 3), 256, 0, stream>>>(
      x16, (const unsigned*)w16, qb, kb, vb,
      (unsigned char*)(ws + OFF_Q4), (_Float16*)(ws + OFF_V16));

  attn_kernel<<<dim3(S_LEN / (16 * 8), 32), 256, 0, stream>>>(
      (const unsigned char*)(ws + OFF_Q4), (const unsigned char*)(ws + OFF_K4),
      (const unsigned*)(ws + OFF_V16), (_Float16*)(ws + OFF_AT16));

  out_gemm_kernel<<<dim3(D_LEN / 64, M_LEN / 128), 256, 0, stream>>>(
      (const _Float16*)(ws + OFF_AT16), (const unsigned*)ow16, ob, (float*)d_out);
}
